// VIG_66520453480970
// MI455X (gfx1250) — matrix operand order repaired and checked by reading, not compiled, not run on hardware
//
#include <hip/hip_runtime.h>
#include <hip/hip_bf16.h>
#include <stdint.h>

#define BB   16
#define NN   2048
#define CC   192
#define EE   256
#define NBLK 12
#define BN_RS 0.9999950000374997f   /* rsqrt(1 + 1e-5) */

typedef __attribute__((ext_vector_type(16))) __bf16 v16bf;
typedef __attribute__((ext_vector_type(8)))  float  v8f;

union Frag { v16bf bf; uint4 q[2]; };

static __device__ __forceinline__ unsigned short f2bf(float f) {
  unsigned int u = __float_as_uint(f);
  u += 0x7FFFu + ((u >> 16) & 1u);      // round-to-nearest-even
  return (unsigned short)(u >> 16);
}

// ---------------------------------------------------------------------------
// Generic bf16 WMMA GEMM:  out[m][o] = sum_k A[m][k] * W[o][k]   (+ epilogue)
// A: (M,K) bf16 row-major, W: (O,K) bf16 row-major (i.e. B matrix = W^T).
// Block: 256 thr = 8 waves -> 64(M) x 256(O) tile; wave -> 32(M) x 64(O):
// each B fragment is reused by two A fragments (8 WMMAs / 6 frag loads).
// Epilogue: +bias, eval-BN (y*g*rsqrt(1+eps)+b), +residual, ReLU; writes f32
// and/or bf16 outputs. Batch strides allow the per-batch Gram matrix (X·X^T).
// ---------------------------------------------------------------------------
__global__ __launch_bounds__(256)
void gemm_bf16_wmma(const unsigned short* __restrict__ A, long long sAb, int lda,
                    const unsigned short* __restrict__ Bw, long long sBb, int ldb,
                    int M, int O, int K,
                    const float* __restrict__ bias,
                    const float* __restrict__ bng, const float* __restrict__ bnb,
                    const float* __restrict__ res, long long sRb,
                    int relu,
                    float* __restrict__ outF, long long sFb,
                    unsigned short* __restrict__ outB, long long sBob)
{
  const int lane = threadIdx.x & 31;
  const int wid  = threadIdx.x >> 5;
  const int mW = blockIdx.x * 64 + (wid & 1) * 32;       // 32 rows per wave
  const int oW = blockIdx.y * 256 + (wid >> 1) * 64;     // 64 cols per wave
  if (mW >= M || oW >= O) return;
  const int bz = blockIdx.z;

  const int mLane = lane & 15;
  const int hi    = lane >> 4;                 // 0: lanes 0-15, 1: lanes 16-31
  const unsigned short* Arow0 = A  + (long long)bz * sAb
                              + (long long)(mW + mLane) * lda + hi * 8;
  const unsigned short* Arow1 = Arow0 + (long long)16 * lda;
  const unsigned short* Brow0 = Bw + (long long)bz * sBb
                              + (long long)(oW + mLane) * ldb + hi * 8;

  v8f acc[2][4] = {};
  for (int k0 = 0; k0 < K; k0 += 32) {
    Frag a0, a1;
    a0.q[0] = *reinterpret_cast<const uint4*>(Arow0 + k0);        // K = hi*8 + 0..7
    a0.q[1] = *reinterpret_cast<const uint4*>(Arow0 + k0 + 16);   // K = hi*8 + 16..23
    a1.q[0] = *reinterpret_cast<const uint4*>(Arow1 + k0);
    a1.q[1] = *reinterpret_cast<const uint4*>(Arow1 + k0 + 16);
#pragma unroll
    for (int t = 0; t < 4; ++t) {
      const unsigned short* Brow = Brow0 + (long long)(t * 16) * ldb + k0;
      Frag b;
      b.q[0] = *reinterpret_cast<const uint4*>(Brow);             // K = hi*8 + 0..7
      b.q[1] = *reinterpret_cast<const uint4*>(Brow + 16);        // K = hi*8 + 16..23
      acc[0][t] = __builtin_amdgcn_wmma_f32_16x16x32_bf16(
          false, a0.bf, false, b.bf, (short)0, acc[0][t], false, false);
      acc[1][t] = __builtin_amdgcn_wmma_f32_16x16x32_bf16(
          false, a1.bf, false, b.bf, (short)0, acc[1][t], false, false);
    }
  }

#pragma unroll
  for (int t = 0; t < 4; ++t) {
    const int o = oW + t * 16 + mLane;
    float bs = bias ? bias[o] : 0.0f;
    float sc = 1.0f, sh = 0.0f;
    if (bng) { sc = bng[o] * BN_RS; sh = bnb[o]; }
#pragma unroll
    for (int a = 0; a < 2; ++a) {
      const int mBase = mW + a * 16 + hi * 8;
#pragma unroll
      for (int r = 0; r < 8; ++r) {
        const int m = mBase + r;
        float v = acc[a][t][r] + bs;
        if (bng) v = v * sc + sh;
        if (res)  v += res[(long long)bz * sRb + (long long)m * O + o];
        if (relu) v = v > 0.0f ? v : 0.0f;
        const long long oi = (long long)m * O + o;
        if (outF) outF[(long long)bz * sFb  + oi] = v;
        if (outB) outB[(long long)bz * sBob + oi] = f2bf(v);
      }
    }
  }
}

// ---------------------------------------------------------------------------
// Top-k (smallest) of dist_row[j] = sq[j] - 2*dot[i][j], with dilated slicing.
// One wave per row; row cached in LDS; k*dil iterative argmin + shfl reduce.
// ---------------------------------------------------------------------------
__global__ __launch_bounds__(128)
void topk_kernel(const float* __restrict__ dot, const float* __restrict__ sq,
                 int* __restrict__ idxOut, int rowStart, int rowsPerChunk,
                 int k, int dil)
{
  __shared__ float lds[4 * NN];
  const int lane = threadIdx.x & 31;
  const int wid  = threadIdx.x >> 5;
  const int r = blockIdx.x * 4 + wid;           // local row in chunk
  const int b = blockIdx.y;
  float* row = lds + wid * NN;
  const float* src = dot + ((long long)b * rowsPerChunk + r) * NN;
  const float* sqb = sq + (long long)b * NN;
  for (int j = lane; j < NN; j += 32)
    row[j] = sqb[j] - 2.0f * src[j];

  const int kd = k * dil;
  const long long obase = ((long long)b * NN + (rowStart + r)) * (long long)k;
  for (int it = 0; it < kd; ++it) {
    float bv = 3.4e38f; int bi = 0;
    for (int j = lane; j < NN; j += 32) {
      float v = row[j];
      if (v < bv) { bv = v; bi = j; }
    }
#pragma unroll
    for (int off = 16; off > 0; off >>= 1) {
      float ov = __shfl_xor(bv, off, 32);
      int   oi = __shfl_xor(bi, off, 32);
      if (ov < bv || (ov == bv && oi < bi)) { bv = ov; bi = oi; }
    }
    if (lane == 0) {
      if ((it % dil) == 0) idxOut[obase + it / dil] = bi;
      row[bi] = 3.4e38f;                         // remove extracted element
    }
  }
}

// ---------------------------------------------------------------------------
// MRConv gather+max: y[n][2c]=x[n][c], y[n][2c+1]=max_j(x[idx[j]][c]-x[n][c])
// Output interleaved (B,N,2C) in bf16 (feeds next GEMM directly).
// ---------------------------------------------------------------------------
__global__ void mrconv_kernel(const float* __restrict__ x, const int* __restrict__ idx,
                              int k, unsigned short* __restrict__ yb)
{
  long long gid = (long long)blockIdx.x * blockDim.x + threadIdx.x;
  if (gid >= (long long)BB * NN * CC) return;
  const int c = (int)(gid % CC);
  const long long bn = gid / CC;
  const int b = (int)(bn / NN);
  const float xv = x[gid];
  const int* ip = idx + bn * k;
  const float* xb = x + (long long)b * NN * CC;
  float xm = -3.4e38f;
  for (int j = 0; j < k; ++j) {
    float d = xb[(long long)ip[j] * CC + c] - xv;
    if (d > xm) xm = d;
  }
  unsigned short* yp = yb + bn * (2 * CC) + 2 * c;
  yp[0] = f2bf(xv);
  yp[1] = f2bf(xm);
}

// --- head helpers: points are C=3, padded to K=32 for the WMMA path --------
__global__ void pad_points_kernel(const float* __restrict__ p, unsigned short* __restrict__ out)
{
  long long bn = (long long)blockIdx.x * blockDim.x + threadIdx.x;
  if (bn >= (long long)BB * NN) return;
  __align__(16) unsigned short v[32];
#pragma unroll
  for (int i = 0; i < 32; ++i) v[i] = 0;
  for (int c = 0; c < 3; ++c) v[c] = f2bf(p[bn * 3 + c]);
  uint4* o = (uint4*)(out + bn * 32);
  const uint4* s = (const uint4*)v;
  o[0] = s[0]; o[1] = s[1]; o[2] = s[2]; o[3] = s[3];
}

__global__ void head_mrconv_kernel(const float* __restrict__ p, const int* __restrict__ idx,
                                   unsigned short* __restrict__ out /* (B,N,32) bf16 */)
{
  long long bn = (long long)blockIdx.x * blockDim.x + threadIdx.x;
  if (bn >= (long long)BB * NN) return;
  const int b = (int)(bn / NN);
  const float* pb = p + (long long)b * NN * 3;
  float xv[3], xm[3];
  for (int c = 0; c < 3; ++c) { xv[c] = p[bn * 3 + c]; xm[c] = -3.4e38f; }
  const int* ip = idx + bn * 9;
  for (int j = 0; j < 9; ++j) {
    const float* xj = pb + (long long)ip[j] * 3;
    for (int c = 0; c < 3; ++c) {
      float d = xj[c] - xv[c];
      if (d > xm[c]) xm[c] = d;
    }
  }
  __align__(16) unsigned short v[32];
#pragma unroll
  for (int i = 0; i < 32; ++i) v[i] = 0;
  for (int c = 0; c < 3; ++c) { v[2 * c] = f2bf(xv[c]); v[2 * c + 1] = f2bf(xm[c]); }
  uint4* o = (uint4*)(out + bn * 32);
  const uint4* s = (const uint4*)v;
  o[0] = s[0]; o[1] = s[1]; o[2] = s[2]; o[3] = s[3];
}

__global__ void rowsq_kernel(const float* __restrict__ x, int C, float* __restrict__ sq)
{
  long long bn = (long long)blockIdx.x * blockDim.x + threadIdx.x;
  if (bn >= (long long)BB * NN) return;
  const float* r = x + bn * C;
  float s = 0.0f;
  for (int c = 0; c < C; ++c) s += r[c] * r[c];
  sq[bn] = s;
}

__global__ void cvt_kernel(const float* __restrict__ in, unsigned short* __restrict__ out,
                           long long n)
{
  long long i = (long long)blockIdx.x * blockDim.x + threadIdx.x;
  if (i < n) out[i] = f2bf(in[i]);
}

__global__ void pad_headw_kernel(const float* __restrict__ w, unsigned short* __restrict__ out)
{
  int i = blockIdx.x * blockDim.x + threadIdx.x;
  if (i >= CC * 32) return;
  int row = i >> 5, col = i & 31;
  out[i] = (col < 6) ? f2bf(w[row * 6 + col]) : (unsigned short)0;
}

__global__ void pool_kernel(const float* __restrict__ x, float* __restrict__ pooled)
{
  int i = blockIdx.x * blockDim.x + threadIdx.x;
  if (i >= BB * CC) return;
  int b = i / CC, c = i % CC;
  const float* xb = x + (long long)b * NN * CC + c;
  float s = 0.0f;
  for (int n = 0; n < NN; ++n) s += xb[(long long)n * CC];
  pooled[i] = s * (1.0f / NN);
}

__global__ void pred_kernel(const float* __restrict__ pooled, const float* __restrict__ w,
                            const float* __restrict__ bias, float* __restrict__ out)
{
  int i = blockIdx.x * blockDim.x + threadIdx.x;
  if (i >= BB * EE) return;
  int b = i / EE, e = i % EE;
  const float* pw = w + (long long)e * CC;
  const float* pv = pooled + (long long)b * CC;
  float s = bias[e];
  for (int c = 0; c < CC; ++c) s += pv[c] * pw[c];
  out[i] = s;
}

// ---------------------------------------------------------------------------
extern "C" void kernel_launch(void* const* d_in, const int* in_sizes, int n_in,
                              void* d_out, int out_size, void* d_ws, size_t ws_size,
                              hipStream_t stream)
{
  static const int KNN[NBLK] = {9, 9, 10, 11, 12, 13, 13, 14, 15, 16, 17, 18};
  static const int DIL[NBLK] = {1, 1, 1, 1, 2, 2, 2, 2, 3, 3, 3, 3};

  const float* points  = (const float*)d_in[0];
  const float* head_w  = (const float*)d_in[1];
  const float* head_g  = (const float*)d_in[2];
  const float* head_b  = (const float*)d_in[3];
  const float* g1_w    = (const float*)d_in[4];
  const float* g1_b    = (const float*)d_in[5];
  const float* g1_g    = (const float*)d_in[6];
  const float* g1_bb   = (const float*)d_in[7];
  const float* mr_w    = (const float*)d_in[8];
  const float* mr_b    = (const float*)d_in[9];
  const float* mr_g    = (const float*)d_in[10];
  const float* mr_bb   = (const float*)d_in[11];
  const float* g2_w    = (const float*)d_in[12];
  const float* g2_b    = (const float*)d_in[13];
  const float* g2_g    = (const float*)d_in[14];
  const float* g2_bb   = (const float*)d_in[15];
  const float* f1_w    = (const float*)d_in[16];
  const float* f1_b    = (const float*)d_in[17];
  const float* f1_g    = (const float*)d_in[18];
  const float* f1_bb   = (const float*)d_in[19];
  const float* f2_w    = (const float*)d_in[20];
  const float* f2_b    = (const float*)d_in[21];
  const float* f2_g    = (const float*)d_in[22];
  const float* f2_bb   = (const float*)d_in[23];
  const float* pred_w  = (const float*)d_in[24];
  const float* pred_b  = (const float*)d_in[25];

  char* ws = (char*)d_ws;
  size_t off = 0;
  auto alloc = [&](size_t bytes) -> char* {
    char* p = ws + off;
    off = (off + bytes + 255) & ~(size_t)255;
    return p;
  };

  const long long MTOK = (long long)BB * NN;       // 32768 tokens
  unsigned short* wg1   = (unsigned short*)alloc((size_t)NBLK * CC * CC * 2);
  unsigned short* wmr   = (unsigned short*)alloc((size_t)NBLK * 2 * CC * 2 * CC * 2);
  unsigned short* wg2   = (unsigned short*)alloc((size_t)NBLK * CC * 2 * CC * 2);
  unsigned short* wf1   = (unsigned short*)alloc((size_t)NBLK * 4 * CC * CC * 2);
  unsigned short* wf2   = (unsigned short*)alloc((size_t)NBLK * CC * 4 * CC * 2);
  unsigned short* whead = (unsigned short*)alloc((size_t)CC * 32 * 2);
  float*          xA    = (float*)alloc((size_t)MTOK * CC * 4);
  float*          xB    = (float*)alloc((size_t)MTOK * CC * 4);
  float*          hF    = (float*)alloc((size_t)MTOK * CC * 4);
  unsigned short* bfX   = (unsigned short*)alloc((size_t)MTOK * CC * 2);
  unsigned short* bfY   = (unsigned short*)alloc((size_t)MTOK * 2 * CC * 2);
  unsigned short* bfH   = (unsigned short*)alloc((size_t)MTOK * 4 * CC * 2);
  float*          sqv   = (float*)alloc((size_t)BB * NN * 4);
  int*            idxb  = (int*)alloc((size_t)BB * NN * 18 * 4);
  float*          dist  = (float*)alloc((size_t)BB * 128 * NN * 4);
  unsigned short* ppad  = (unsigned short*)alloc((size_t)BB * NN * 32 * 2);
  unsigned short* hpad  = (unsigned short*)alloc((size_t)BB * NN * 32 * 2);
  float*          pooled= (float*)alloc((size_t)BB * CC * 4);
  (void)ws_size; (void)in_sizes; (void)n_in; (void)out_size;

  auto gemm = [&](const unsigned short* A, long long sAb, int lda,
                  const unsigned short* Bw, long long sBb, int ldb,
                  int M, int O, int K,
                  const float* bias, const float* bng, const float* bnb,
                  const float* res, long long sRb, int relu,
                  float* outF, long long sFb, unsigned short* outB, long long sBob,
                  int nbatch) {
    dim3 g((unsigned)(M / 64), (unsigned)((O + 255) / 256), (unsigned)nbatch);
    gemm_bf16_wmma<<<g, dim3(256), 0, stream>>>(A, sAb, lda, Bw, sBb, ldb, M, O, K,
                                                bias, bng, bnb, res, sRb, relu,
                                                outF, sFb, outB, sBob);
  };

  // Dense kNN: Gram matrix via WMMA (128-row chunks) + wave top-k.
  auto knn = [&](const unsigned short* xb16, int K, const float* xf, int Cfeat,
                 int k, int dil) {
    rowsq_kernel<<<(BB * NN + 255) / 256, 256, 0, stream>>>(xf, Cfeat, sqv);
    for (int rc = 0; rc < NN / 128; ++rc) {
      gemm(xb16 + (long long)rc * 128 * K, (long long)NN * K, K,
           xb16, (long long)NN * K, K,
           128, NN, K,
           nullptr, nullptr, nullptr, nullptr, 0, 0,
           dist, (long long)128 * NN, nullptr, 0, BB);
      topk_kernel<<<dim3(32, BB), 128, 0, stream>>>(dist, sqv, idxb, rc * 128, 128, k, dil);
    }
  };

  // ---- weights -> bf16 (deterministic, every call) ----
  auto cvt = [&](const float* in, unsigned short* out, long long n) {
    cvt_kernel<<<(unsigned)((n + 255) / 256), 256, 0, stream>>>(in, out, n);
  };
  cvt(g1_w, wg1, (long long)NBLK * CC * CC);
  cvt(mr_w, wmr, (long long)NBLK * 2 * CC * 2 * CC);
  cvt(g2_w, wg2, (long long)NBLK * CC * 2 * CC);
  cvt(f1_w, wf1, (long long)NBLK * 4 * CC * CC);
  cvt(f2_w, wf2, (long long)NBLK * CC * 4 * CC);
  pad_headw_kernel<<<(CC * 32 + 255) / 256, 256, 0, stream>>>(head_w, whead);

  // ---- head: kNN on points (K padded 3->32) + MRConv (6->C) ----
  pad_points_kernel<<<(BB * NN + 255) / 256, 256, 0, stream>>>(points, ppad);
  knn(ppad, 32, points, 3, 9, 1);
  head_mrconv_kernel<<<(BB * NN + 255) / 256, 256, 0, stream>>>(points, idxb, hpad);
  gemm(hpad, 0, 32, whead, 0, 32, (int)MTOK, CC, 32,
       nullptr, head_g, head_b, nullptr, 0, 1, xA, 0, bfX, 0, 1);

  float* xcur = xA; float* xnext = xB;
  for (int i = 0; i < NBLK; ++i) {
    // Grapher fc1: C->C, bias+BN
    gemm(bfX, 0, CC, wg1 + (long long)i * CC * CC, 0, CC, (int)MTOK, CC, CC,
         g1_b + i * CC, g1_g + i * CC, g1_bb + i * CC, nullptr, 0, 0,
         hF, 0, bfH, 0, 1);
    // dynamic kNN on h
    knn(bfH, CC, hF, CC, KNN[i], DIL[i]);
    // MRConv gather/max -> interleaved (B,N,2C) bf16
    mrconv_kernel<<<(unsigned)(((long long)BB * NN * CC + 255) / 256), 256, 0, stream>>>(
        hF, idxb, KNN[i], bfY);
    // MRConv GEMM: 2C->2C, bias+BN+ReLU
    gemm(bfY, 0, 2 * CC, wmr + (long long)i * 2 * CC * 2 * CC, 0, 2 * CC,
         (int)MTOK, 2 * CC, 2 * CC,
         mr_b + i * 2 * CC, mr_g + i * 2 * CC, mr_bb + i * 2 * CC, nullptr, 0, 1,
         nullptr, 0, bfH, 0, 1);
    // fc2: 2C->C, bias+BN, +residual
    gemm(bfH, 0, 2 * CC, wg2 + (long long)i * CC * 2 * CC, 0, 2 * CC,
         (int)MTOK, CC, 2 * CC,
         g2_b + i * CC, g2_g + i * CC, g2_bb + i * CC, xcur, 0, 0,
         xnext, 0, bfX, 0, 1);
    { float* t = xcur; xcur = xnext; xnext = t; }
    // FFN fc1: C->4C, bias+BN+ReLU
    gemm(bfX, 0, CC, wf1 + (long long)i * 4 * CC * CC, 0, CC,
         (int)MTOK, 4 * CC, CC,
         f1_b + i * 4 * CC, f1_g + i * 4 * CC, f1_bb + i * 4 * CC, nullptr, 0, 1,
         nullptr, 0, bfH, 0, 1);
    // FFN fc2: 4C->C, bias+BN, +residual
    gemm(bfH, 0, 4 * CC, wf2 + (long long)i * CC * 4 * CC, 0, 4 * CC,
         (int)MTOK, CC, 4 * CC,
         f2_b + i * CC, f2_g + i * CC, f2_bb + i * CC, xcur, 0, 0,
         xnext, 0, bfX, 0, 1);
    { float* t = xcur; xcur = xnext; xnext = t; }
  }

  pool_kernel<<<(BB * CC + 255) / 256, 256, 0, stream>>>(xcur, pooled);
  pred_kernel<<<(BB * EE + 255) / 256, 256, 0, stream>>>(pooled, pred_w, pred_b,
                                                         (float*)d_out);
}
